// SubgraphEncoder_45045617000801
// MI455X (gfx1250) — compile-verified, matrix operand order checked by reading, not run on hardware
//
#include <hip/hip_runtime.h>

typedef __attribute__((ext_vector_type(16))) __bf16 v16bf;
typedef __attribute__((ext_vector_type(8)))  __bf16 v8bf;
typedef __attribute__((ext_vector_type(8)))  float  v8f;
typedef __attribute__((ext_vector_type(4)))  float  v4f;

#define N_NODES  50000
#define N_EDGES  800000
#define DIM_IN   128
#define DIM_H    512
#define DIM_OUT  256
#define N_GRAPHS 512

// ---------------------------------------------------------------------------
// 1) agg = x  (GIN eps = 0 -> h0 = x + sum_{neighbors} x[src])
// ---------------------------------------------------------------------------
__global__ void k_copy_x(const float* __restrict__ x, float* __restrict__ agg) {
  int i = blockIdx.x * blockDim.x + threadIdx.x;           // float4 granules
  if (i < (N_NODES * DIM_IN) / 4)
    ((v4f*)agg)[i] = ((const v4f*)x)[i];
}

// one thread per (edge, 4-feature chunk): float4 load + 4 atomic f32 adds
__global__ void k_scatter_edges(const float* __restrict__ x,
                                const int* __restrict__ ei,
                                float* __restrict__ agg) {
  int t = blockIdx.x * blockDim.x + threadIdx.x;
  if (t >= N_EDGES * 32) return;
  int e = t >> 5;
  int c = (t & 31) * 4;
  int src = ei[e];
  int dst = ei[N_EDGES + e];
  src = (src < 0) ? 0 : (src >= N_NODES ? N_NODES - 1 : src);
  dst = (dst < 0) ? 0 : (dst >= N_NODES ? N_NODES - 1 : dst);
  v4f v = *(const v4f*)(x + (size_t)src * DIM_IN + c);
  float* a = agg + (size_t)dst * DIM_IN + c;
  atomicAdd(a + 0, v[0]);
  atomicAdd(a + 1, v[1]);
  atomicAdd(a + 2, v[2]);
  atomicAdd(a + 3, v[3]);
}

// ---------------------------------------------------------------------------
// 2) pack weight [K, Dout] f32 (row major) into bf16 WMMA B-fragments.
// Fragment (nt, kt) = 32x16 tile; lane L holds column N = L&15, 16 bf16 with
// K = kt*32 + (L>=16 ? 8 : 0) + (e < 8 ? e : e + 8)   (ISA 16-bit layout)
// Stored contiguously: frag*512 + lane*16 elements (1024 B per fragment).
// ---------------------------------------------------------------------------
__global__ void k_pack_weight(const float* __restrict__ W,
                              __bf16* __restrict__ Wp,
                              int K, int Dout) {
  const int gwave = (blockIdx.x * blockDim.x + threadIdx.x) >> 5;
  const int lane  = threadIdx.x & 31;
  const int KT    = K >> 5;
  const int nfrag = (Dout >> 4) * KT;
  if (gwave >= nfrag) return;
  const int nt = gwave / KT;
  const int kt = gwave % KT;
  const int nc = nt * 16 + (lane & 15);
  const int ha = (lane >> 4) * 8;
  __bf16* dst = Wp + (size_t)gwave * 512 + lane * 16;
#pragma unroll
  for (int e = 0; e < 16; ++e) {
    int k = kt * 32 + ha + (e < 8 ? e : e + 8);
    dst[e] = (__bf16)W[(size_t)k * Dout + nc];
  }
}

// ---------------------------------------------------------------------------
// 3) WMMA GEMM layer: Hout[N, Dout] = act(A[N, K] @ W + b), bf16 in/out,
//    f32 accumulate. 8 waves/block: block tile 64 rows x 256 cols,
//    wave tile 32 rows x 64 cols = 2 A-frags x 4 B-frags -> 8 WMMAs per
//    K-step from 12 b128 loads (1.5 loads/WMMA). K compile-time -> fully
//    unrolled, immediate-offset loads, register double-buffering.
//    ACT: 0 none, 1 relu, 2 leaky(1.5)
// ---------------------------------------------------------------------------
template <int ACT, bool AF32, int K>
__global__ __launch_bounds__(256)
void k_gin_gemm(const void* __restrict__ Ain,
                const __bf16* __restrict__ Wp,
                const float* __restrict__ bias,
                __bf16* __restrict__ Hout,
                int Dout) {
  constexpr int KT = K / 32;
  const int lane  = threadIdx.x & 31;
  const int wave  = threadIdx.x >> 5;
  const int half  = lane >> 4;
  const int m     = lane & 15;
  const int mbase = blockIdx.x * 64 + (wave & 1) * 32;
  const int nt0   = blockIdx.y * 16 + (wave >> 1) * 4;

  int arow[2];
  arow[0] = mbase + m;
  arow[1] = mbase + 16 + m;
  if (arow[0] >= N_NODES) arow[0] = N_NODES - 1;   // clamp; stores guarded
  if (arow[1] >= N_NODES) arow[1] = N_NODES - 1;

  // fixed base pointers; per-kt deltas are compile-time immediates
  const float*  apf[2];
  const __bf16* apb[2];
#pragma unroll
  for (int i = 0; i < 2; ++i) {
    apf[i] = (const float*)Ain  + (size_t)arow[i] * K + half * 8;
    apb[i] = (const __bf16*)Ain + (size_t)arow[i] * K + half * 8;
  }
  const __bf16* bbase[4];
#pragma unroll
  for (int n = 0; n < 4; ++n)
    bbase[n] = Wp + (size_t)(nt0 + n) * KT * 512 + lane * 16;

  auto load_a = [&](int i, int kt) -> v16bf {
    v16bf a;
    if constexpr (AF32) {
      const float* ap = apf[i] + kt * 32;
      v4f l0 = *(const v4f*)(ap);
      v4f l1 = *(const v4f*)(ap + 4);
      v4f h0 = *(const v4f*)(ap + 16);
      v4f h1 = *(const v4f*)(ap + 20);
#pragma unroll
      for (int q = 0; q < 4; ++q) {
        a[q]      = (__bf16)l0[q];
        a[q + 4]  = (__bf16)l1[q];
        a[q + 8]  = (__bf16)h0[q];
        a[q + 12] = (__bf16)h1[q];
      }
    } else {
      const __bf16* ap = apb[i] + kt * 32;
      v8bf lo = *(const v8bf*)(ap);
      v8bf hi = *(const v8bf*)(ap + 16);
#pragma unroll
      for (int q = 0; q < 8; ++q) { a[q] = lo[q]; a[q + 8] = hi[q]; }
    }
    return a;
  };
  auto load_b = [&](int n, int kt) -> v16bf {
    const __bf16* bp = bbase[n] + kt * 512;
    v8bf lo = *(const v8bf*)(bp);
    v8bf hi = *(const v8bf*)(bp + 8);
    v16bf b;
#pragma unroll
    for (int q = 0; q < 8; ++q) { b[q] = lo[q]; b[q + 8] = hi[q]; }
    return b;
  };

  v8f acc[2][4] = {};
  v16bf a_cur[2], b_cur[4];
#pragma unroll
  for (int i = 0; i < 2; ++i) a_cur[i] = load_a(i, 0);
#pragma unroll
  for (int n = 0; n < 4; ++n) b_cur[n] = load_b(n, 0);

#pragma unroll
  for (int kt = 0; kt < KT; ++kt) {
    v16bf a_nxt[2], b_nxt[4];
    if (kt + 1 < KT) {                    // compile-time resolved (unrolled)
#pragma unroll
      for (int i = 0; i < 2; ++i) a_nxt[i] = load_a(i, kt + 1);
#pragma unroll
      for (int n = 0; n < 4; ++n) b_nxt[n] = load_b(n, kt + 1);
    }
#pragma unroll
    for (int i = 0; i < 2; ++i)
#pragma unroll
      for (int n = 0; n < 4; ++n)
        acc[i][n] = __builtin_amdgcn_wmma_f32_16x16x32_bf16(
            false, a_cur[i], false, b_cur[n], (short)0, acc[i][n], false, false);
    if (kt + 1 < KT) {
#pragma unroll
      for (int i = 0; i < 2; ++i) a_cur[i] = a_nxt[i];
#pragma unroll
      for (int n = 0; n < 4; ++n) b_cur[n] = b_nxt[n];
    }
  }

  // epilogue: bias + activation + bf16 store. D layout: VGPR r, lane L ->
  // row = mbase + i*16 + r + (L>=16 ? 8 : 0), col = (nt0+n)*16 + (L&15)
#pragma unroll
  for (int n = 0; n < 4; ++n) {
    const int nc = (nt0 + n) * 16 + m;
    const float bv = bias[nc];
#pragma unroll
    for (int i = 0; i < 2; ++i) {
#pragma unroll
      for (int r = 0; r < 8; ++r) {
        int row = mbase + i * 16 + r + half * 8;
        float v = acc[i][n][r] + bv;
        if (ACT == 1) v = fmaxf(v, 0.f);
        else if (ACT == 2) v = (v > 0.f) ? v : 1.5f * v;
        if (row < N_NODES) Hout[(size_t)row * Dout + nc] = (__bf16)v;
      }
    }
  }
}

// ---------------------------------------------------------------------------
// 4) mean pool + head
// ---------------------------------------------------------------------------
__global__ void k_zero_f32(float* __restrict__ p, int n) {
  int i = blockIdx.x * blockDim.x + threadIdx.x;
  if (i < n) p[i] = 0.f;
}

__global__ void k_pool_accum(const __bf16* __restrict__ h,
                             const int* __restrict__ batch,
                             float* __restrict__ psum) {
  int t = blockIdx.x * blockDim.x + threadIdx.x;
  if (t >= N_NODES * DIM_H) return;
  int node = t >> 9;
  int g = batch[node];
  g = (g < 0) ? 0 : (g >= N_GRAPHS ? N_GRAPHS - 1 : g);
  atomicAdd(&psum[(size_t)g * DIM_H + (t & 511)], (float)h[t]);
}

__global__ void k_pool_count(const int* __restrict__ batch, float* __restrict__ cnt) {
  int i = blockIdx.x * blockDim.x + threadIdx.x;
  if (i >= N_NODES) return;
  int g = batch[i];
  g = (g < 0) ? 0 : (g >= N_GRAPHS ? N_GRAPHS - 1 : g);
  atomicAdd(&cnt[g], 1.0f);
}

// head: out[g, :] = normalize(pooled[g] @ Wl + bl). One block per graph.
__global__ __launch_bounds__(256)
void k_head_norm(const float* __restrict__ psum,
                 const float* __restrict__ cnt,
                 const float* __restrict__ Wl,
                 const float* __restrict__ bl,
                 float* __restrict__ out) {
  __shared__ float prow[DIM_H];
  __shared__ float red[256];
  const int g = blockIdx.x;
  const int j = threadIdx.x;
  const float inv = 1.0f / fmaxf(cnt[g], 1.0f);
  for (int k = j; k < DIM_H; k += 256) prow[k] = psum[(size_t)g * DIM_H + k] * inv;
  __syncthreads();
  float acc = bl[j];
#pragma unroll 8
  for (int k = 0; k < DIM_H; ++k) acc += prow[k] * Wl[(size_t)k * DIM_OUT + j];
  red[j] = acc * acc;
  __syncthreads();
  for (int s = 128; s > 0; s >>= 1) {
    if (j < s) red[j] += red[j + s];
    __syncthreads();
  }
  float nrm = fmaxf(sqrtf(red[0]), 1e-12f);
  out[(size_t)g * DIM_OUT + j] = acc / nrm;
}

// ---------------------------------------------------------------------------
extern "C" void kernel_launch(void* const* d_in, const int* in_sizes, int n_in,
                              void* d_out, int out_size, void* d_ws, size_t ws_size,
                              hipStream_t stream) {
  const float* x     = (const float*)d_in[0];
  const int*   ei    = (const int*)d_in[1];
  const int*   batch = (const int*)d_in[2];
  const float* W1 = (const float*)d_in[3];  const float* b1 = (const float*)d_in[4];
  const float* W2 = (const float*)d_in[5];  const float* b2 = (const float*)d_in[6];
  const float* W3 = (const float*)d_in[7];  const float* b3 = (const float*)d_in[8];
  const float* W4 = (const float*)d_in[9];  const float* b4 = (const float*)d_in[10];
  const float* Wl = (const float*)d_in[11]; const float* bl = (const float*)d_in[12];
  float* out = (float*)d_out;

  char* ws = (char*)d_ws;
  size_t off = 0;
  auto take = [&](size_t bytes) -> char* {
    char* p = ws + off;
    off = (off + bytes + 255) & ~(size_t)255;
    return p;
  };
  float*  agg  = (float*)take((size_t)N_NODES * DIM_IN * 4);
  __bf16* hA   = (__bf16*)take((size_t)N_NODES * DIM_H * 2);
  __bf16* hB   = (__bf16*)take((size_t)N_NODES * DIM_H * 2);
  __bf16* wp1  = (__bf16*)take((size_t)DIM_IN * DIM_H * 2);
  __bf16* wp2  = (__bf16*)take((size_t)DIM_H * DIM_H * 2);
  __bf16* wp3  = (__bf16*)take((size_t)DIM_H * DIM_H * 2);
  __bf16* wp4  = (__bf16*)take((size_t)DIM_H * DIM_H * 2);
  float*  psum = (float*)take((size_t)N_GRAPHS * DIM_H * 4);
  float*  cnt  = (float*)take((size_t)N_GRAPHS * 4);

  // 1) GIN aggregation
  k_copy_x<<<((N_NODES * DIM_IN / 4) + 255) / 256, 256, 0, stream>>>(x, agg);
  k_scatter_edges<<<(N_EDGES * 32 + 255) / 256, 256, 0, stream>>>(x, ei, agg);

  // 2) pack weights into bf16 WMMA fragments (8 frags / 256-thread block)
  {
    int nf1 = (DIM_H / 16) * (DIM_IN / 32);                  // 128
    int nfh = (DIM_H / 16) * (DIM_H / 32);                   // 512
    k_pack_weight<<<(nf1 * 32 + 255) / 256, 256, 0, stream>>>(W1, wp1, DIM_IN, DIM_H);
    k_pack_weight<<<(nfh * 32 + 255) / 256, 256, 0, stream>>>(W2, wp2, DIM_H, DIM_H);
    k_pack_weight<<<(nfh * 32 + 255) / 256, 256, 0, stream>>>(W3, wp3, DIM_H, DIM_H);
    k_pack_weight<<<(nfh * 32 + 255) / 256, 256, 0, stream>>>(W4, wp4, DIM_H, DIM_H);
  }

  // 3) MLP via WMMA (ping-pong hA/hB); block tile 64x256
  dim3 gg((N_NODES + 63) / 64, DIM_H / 256);                 // 782 x 2
  k_gin_gemm<2, true,  DIM_IN><<<gg, 256, 0, stream>>>(agg, wp1, b1, hA, DIM_H);
  k_gin_gemm<1, false, DIM_H ><<<gg, 256, 0, stream>>>(hA,  wp2, b2, hB, DIM_H);
  k_gin_gemm<1, false, DIM_H ><<<gg, 256, 0, stream>>>(hB,  wp3, b3, hA, DIM_H);
  k_gin_gemm<0, false, DIM_H ><<<gg, 256, 0, stream>>>(hA,  wp4, b4, hB, DIM_H);

  // 4) mean pool + head + normalize
  k_zero_f32<<<(N_GRAPHS * DIM_H + 255) / 256, 256, 0, stream>>>(psum, N_GRAPHS * DIM_H);
  k_zero_f32<<<(N_GRAPHS + 255) / 256, 256, 0, stream>>>(cnt, N_GRAPHS);
  k_pool_accum<<<(N_NODES * DIM_H + 255) / 256, 256, 0, stream>>>(hB, batch, psum);
  k_pool_count<<<(N_NODES + 255) / 256, 256, 0, stream>>>(batch, cnt);
  k_head_norm<<<N_GRAPHS, 256, 0, stream>>>(psum, cnt, Wl, bl, out);
}
